// MultiHeadAttention_T_4818953306886
// MI455X (gfx1250) — matrix operand order repaired and checked by reading, not compiled, not run on hardware
//
#include <hip/hip_runtime.h>

#define BB 8
#define SS 2048
#define DD 64
#define HH 8
#define HD 512
#define NR (BB*SS)   /* 16384 rows */

typedef __attribute__((ext_vector_type(16))) _Float16 v16h;
typedef __attribute__((ext_vector_type(8)))  float    v8f;
typedef __attribute__((ext_vector_type(4)))  unsigned int u32x4;
typedef unsigned int u32;
typedef _Float16 f16;

union AFrag { v16h h; u32 d[8]; };
union BFrag { v16h h; u32x4 q[2]; };
union H16   { u32x4 q[2]; f16 h[16]; };

// k-offset of dword r of an f16 A-fragment (16x32), contiguous order k = 16*hi + 2*r
__device__ __forceinline__ int kmap(int r, int hi) {
    return (r << 1) + (hi << 4);
}
__device__ __forceinline__ v8f zero8() {
    v8f z; for (int i = 0; i < 8; ++i) z[i] = 0.f; return z;
}
__device__ __forceinline__ v8f wmma16(const AFrag& a, const BFrag& b, v8f c) {
    return __builtin_amdgcn_wmma_f32_16x16x32_f16(false, a.h, false, b.h,
                                                  (short)0, c, false, false);
}

// ---------------------------------------------------------------------------
// Kernel 1: QKV projection.  grid (128, 3): x = 128-row tile, y = {Q,K,V}.
// Out: f16 row-major [16384][512]; Q pre-scaled by D^-0.5 = 0.125.
// ---------------------------------------------------------------------------
__global__ __launch_bounds__(256)
void qkv_kernel(const float* __restrict__ X,
                const float* __restrict__ Wq, const float* __restrict__ bq,
                const float* __restrict__ Wk, const float* __restrict__ bk,
                const float* __restrict__ Wv, const float* __restrict__ bv,
                f16* __restrict__ Qh, f16* __restrict__ Kh, f16* __restrict__ Vh)
{
    __shared__ alignas(16) f16 Xs[128 * 64];    // X tile, f16
    __shared__ alignas(16) f16 Ws[256 * 72];    // W^T half: Bt[n][k], stride 72
    const int t = threadIdx.x;
    const int w = t >> 5, lane = t & 31, hi = lane >> 4, lo = lane & 15;
    const int rowbase = blockIdx.x * 128;
    const int which = blockIdx.y;
    const float* W    = which == 0 ? Wq : (which == 1 ? Wk : Wv);
    const float* bias = which == 0 ? bq : (which == 1 ? bk : bv);
    f16* out          = which == 0 ? Qh : (which == 1 ? Kh : Vh);
    const float scale = which == 0 ? 0.125f : 1.0f;

    for (int i = 0; i < 32; ++i) {
        int f = t + i * 256;
        Xs[f] = (f16)X[(size_t)rowbase * DD + f];
    }
    __syncthreads();

    AFrag a0, a1;                                    // rows w*16 .. w*16+15
    {
        const f16* arow = Xs + (w * 16 + lo) * 64;
        for (int r = 0; r < 8; ++r) {
            int k = kmap(r, hi);
            a0.d[r] = *(const u32*)(arow + k);
            a1.d[r] = *(const u32*)(arow + k + 32);
        }
    }
    for (int half = 0; half < 2; ++half) {
        __syncthreads();
        for (int i = 0; i < 64; ++i)                 // Ws[n][k] = W[k][half*256+n]
            Ws[t * 72 + i] = (f16)W[(size_t)i * HD + half * 256 + t];
        __syncthreads();
        for (int nt = 0; nt < 16; ++nt) {
            BFrag b0, b1;
            const f16* brow = Ws + (nt * 16 + lo) * 72 + hi * 16;
            b0.q[0] = *(const u32x4*)(brow);
            b0.q[1] = *(const u32x4*)(brow + 8);
            b1.q[0] = *(const u32x4*)(brow + 32);
            b1.q[1] = *(const u32x4*)(brow + 40);
            v8f c = zero8();
            c = wmma16(a0, b0, c);
            c = wmma16(a1, b1, c);
            int nglob = half * 256 + nt * 16 + lo;
            float bb = bias[nglob];
            size_t base = (size_t)(rowbase + w * 16 + 8 * hi) * HD + nglob;
            for (int r = 0; r < 8; ++r)
                out[base + (size_t)r * HD] = (f16)((c[r] + bb) * scale);
        }
    }
}

// ---------------------------------------------------------------------------
// Kernel 2: flash attention.  grid (16, 64): x = 128-row q tile, y = (b,h).
// 8 waves, each owns one 16-row m-tile; j-chunks of 64.
// ---------------------------------------------------------------------------
__global__ __launch_bounds__(256)
void attn_kernel(const f16* __restrict__ Qh, const f16* __restrict__ Kh,
                 const f16* __restrict__ Vh, f16* __restrict__ Ch)
{
    __shared__ alignas(16) f16 Ks[64 * 72];        // K chunk [j][d], stride 72
    __shared__ alignas(16) f16 Vs[64 * 72];        // V chunk transposed [d][j]
    __shared__ alignas(16) f16 Ps[8 * 16 * 68];    // per-wave P, stride 68
    const int t = threadIdx.x;
    const int w = t >> 5, lane = t & 31, hi = lane >> 4, lo = lane & 15;
    const size_t gbase = (size_t)blockIdx.y * (SS * DD);
    const int q0 = blockIdx.x * 128 + w * 16;

    AFrag a0, a1;                                   // Q fragments (held in regs)
    {
        const f16* qrow = Qh + gbase + (size_t)(q0 + lo) * DD;
        for (int r = 0; r < 8; ++r) {
            int k = kmap(r, hi);
            a0.d[r] = *(const u32*)(qrow + k);
            a1.d[r] = *(const u32*)(qrow + k + 32);
        }
    }
    float mrow[8], lrow[8];
    v8f acc[4];
    for (int r = 0; r < 8; ++r) { mrow[r] = -3.0e38f; lrow[r] = 0.f; }
    for (int dt = 0; dt < 4; ++dt) acc[dt] = zero8();

    const int jl = t >> 2;             // coop-copy row 0..63
    const int d0 = (t & 3) * 16;       // coop-copy col base

    for (int jc = 0; jc < SS; jc += 64) {
        __syncthreads();
        {   // stage K chunk (row-major) and V chunk (transposed) in LDS
            const f16* ksrc = Kh + gbase + (size_t)(jc + jl) * DD + d0;
            *(u32x4*)&Ks[jl * 72 + d0]     = *(const u32x4*)(ksrc);
            *(u32x4*)&Ks[jl * 72 + d0 + 8] = *(const u32x4*)(ksrc + 8);
            H16 vv;
            const f16* vsrc = Vh + gbase + (size_t)(jc + jl) * DD + d0;
            vv.q[0] = *(const u32x4*)(vsrc);
            vv.q[1] = *(const u32x4*)(vsrc + 8);
            for (int i = 0; i < 16; ++i)
                Vs[(d0 + i) * 72 + jl] = vv.h[i];
        }
        __syncthreads();

        v8f s[4];                                   // scores S = (Q*0.125) K^T
        for (int nt = 0; nt < 4; ++nt) {
            BFrag b0, b1;
            const f16* brow = Ks + (nt * 16 + lo) * 72 + hi * 16;
            b0.q[0] = *(const u32x4*)(brow);
            b0.q[1] = *(const u32x4*)(brow + 8);
            b1.q[0] = *(const u32x4*)(brow + 32);
            b1.q[1] = *(const u32x4*)(brow + 40);
            v8f c = zero8();
            c = wmma16(a0, b0, c);
            c = wmma16(a1, b1, c);
            s[nt] = c;
        }
        // online softmax; row (r,hi) lives in one 16-lane half
        for (int r = 0; r < 8; ++r) {
            float cm = fmaxf(fmaxf(s[0][r], s[1][r]), fmaxf(s[2][r], s[3][r]));
            cm = fmaxf(cm, __shfl_xor(cm, 1));
            cm = fmaxf(cm, __shfl_xor(cm, 2));
            cm = fmaxf(cm, __shfl_xor(cm, 4));
            cm = fmaxf(cm, __shfl_xor(cm, 8));
            float mnew = fmaxf(mrow[r], cm);
            float corr = __expf(mrow[r] - mnew);
            float p0 = __expf(s[0][r] - mnew);
            float p1 = __expf(s[1][r] - mnew);
            float p2 = __expf(s[2][r] - mnew);
            float p3 = __expf(s[3][r] - mnew);
            s[0][r] = p0; s[1][r] = p1; s[2][r] = p2; s[3][r] = p3;
            float rs = (p0 + p1) + (p2 + p3);
            rs += __shfl_xor(rs, 1);
            rs += __shfl_xor(rs, 2);
            rs += __shfl_xor(rs, 4);
            rs += __shfl_xor(rs, 8);
            lrow[r] = lrow[r] * corr + rs;
            mrow[r] = mnew;
            for (int dt = 0; dt < 4; ++dt) acc[dt][r] *= corr;
        }
        // relayout P: D-fragment -> wave-private LDS -> A-fragment
        f16* pw = Ps + w * (16 * 68);
        for (int nt = 0; nt < 4; ++nt)
            for (int r = 0; r < 8; ++r)
                pw[(r + 8 * hi) * 68 + nt * 16 + lo] = (f16)s[nt][r];
        AFrag pa0, pa1;
        {
            const f16* prow = pw + lo * 68;
            for (int r = 0; r < 8; ++r) {
                int k = kmap(r, hi);
                pa0.d[r] = *(const u32*)(prow + k);
                pa1.d[r] = *(const u32*)(prow + k + 32);
            }
        }
        for (int dt = 0; dt < 4; ++dt) {            // ctx += P V
            BFrag b0, b1;
            const f16* brow = Vs + (dt * 16 + lo) * 72 + hi * 16;
            b0.q[0] = *(const u32x4*)(brow);
            b0.q[1] = *(const u32x4*)(brow + 8);
            b1.q[0] = *(const u32x4*)(brow + 32);
            b1.q[1] = *(const u32x4*)(brow + 40);
            acc[dt] = wmma16(pa0, b0, acc[dt]);
            acc[dt] = wmma16(pa1, b1, acc[dt]);
        }
    }
    float inv[8];
    for (int r = 0; r < 8; ++r) inv[r] = 1.0f / lrow[r];
    for (int dt = 0; dt < 4; ++dt) {
        size_t base = gbase + (size_t)(q0 + 8 * hi) * DD + dt * 16 + lo;
        for (int r = 0; r < 8; ++r)
            Ch[base + (size_t)r * DD] = (f16)(acc[dt][r] * inv[r]);
    }
}

// ---------------------------------------------------------------------------
// Kernel 3: out-projection [16384x512]x[512x64] + bias + residual + LayerNorm
// ---------------------------------------------------------------------------
__global__ __launch_bounds__(256)
void oln_kernel(const f16* __restrict__ Ch, const float* __restrict__ Wo,
                const float* __restrict__ bo, const float* __restrict__ X,
                const float* __restrict__ gamma, const float* __restrict__ beta,
                float* __restrict__ out)
{
    __shared__ alignas(16) f16 Ws[64 * 264];       // Wo^T K-half, stride 264
    const int t = threadIdx.x;
    const int w = t >> 5, lane = t & 31, hi = lane >> 4, lo = lane & 15;
    const int rowbase = blockIdx.x * 128;
    const int nl = t & 63, kq = t >> 6;

    v8f acc[4];
    for (int nt = 0; nt < 4; ++nt) acc[nt] = zero8();

    for (int kh = 0; kh < 2; ++kh) {
        __syncthreads();
        for (int i = 0; i < 64; ++i) {
            int kl = i * 4 + kq;
            Ws[nl * 264 + kl] = (f16)Wo[(size_t)(kh * 256 + kl) * 64 + nl];
        }
        __syncthreads();
        for (int ks = 0; ks < 8; ++ks) {
            AFrag a;
            const f16* arow = Ch + (size_t)(rowbase + w * 16 + lo) * HD
                                 + kh * 256 + ks * 32;
            for (int r = 0; r < 8; ++r)
                a.d[r] = *(const u32*)(arow + kmap(r, hi));
            for (int nt = 0; nt < 4; ++nt) {
                BFrag b;
                const f16* brow = Ws + (nt * 16 + lo) * 264 + ks * 32 + hi * 16;
                b.q[0] = *(const u32x4*)(brow);
                b.q[1] = *(const u32x4*)(brow + 8);
                acc[nt] = wmma16(a, b, acc[nt]);
            }
        }
    }
    float bov[4], gv[4], bev[4];
    for (int nt = 0; nt < 4; ++nt) {
        int n = nt * 16 + lo;
        bov[nt] = bo[n]; gv[nt] = gamma[n]; bev[nt] = beta[n];
    }
    float y[4];
    for (int r = 0; r < 8; ++r) {
        int row = rowbase + w * 16 + 8 * hi + r;
        float sum = 0.f, sq = 0.f;
        for (int nt = 0; nt < 4; ++nt) {
            int n = nt * 16 + lo;
            float v = acc[nt][r] + bov[nt] + X[(size_t)row * DD + n];
            y[nt] = v;
            sum += v;
        }
        sum += __shfl_xor(sum, 1); sum += __shfl_xor(sum, 2);
        sum += __shfl_xor(sum, 4); sum += __shfl_xor(sum, 8);
        float mean = sum * (1.0f / 64.0f);
        for (int nt = 0; nt < 4; ++nt) { float d = y[nt] - mean; sq += d * d; }
        sq += __shfl_xor(sq, 1); sq += __shfl_xor(sq, 2);
        sq += __shfl_xor(sq, 4); sq += __shfl_xor(sq, 8);
        float rstd = rsqrtf(sq * (1.0f / 64.0f) + 1e-5f);
        for (int nt = 0; nt < 4; ++nt) {
            int n = nt * 16 + lo;
            out[(size_t)row * DD + n] = (y[nt] - mean) * rstd * gv[nt] + bev[nt];
        }
    }
}

extern "C" void kernel_launch(void* const* d_in, const int* in_sizes, int n_in,
                              void* d_out, int out_size, void* d_ws, size_t ws_size,
                              hipStream_t stream) {
    (void)in_sizes; (void)n_in; (void)out_size; (void)ws_size;
    const float* x     = (const float*)d_in[0];
    const float* Wq    = (const float*)d_in[1];
    const float* bq    = (const float*)d_in[2];
    const float* Wk    = (const float*)d_in[3];
    const float* bk    = (const float*)d_in[4];
    const float* Wv    = (const float*)d_in[5];
    const float* bv    = (const float*)d_in[6];
    const float* Wo    = (const float*)d_in[7];
    const float* bo    = (const float*)d_in[8];
    const float* gamma = (const float*)d_in[9];
    const float* beta  = (const float*)d_in[10];
    float* out = (float*)d_out;

    f16* Qh = (f16*)d_ws;                    // 16 MB each, 64 MB total scratch
    f16* Kh = Qh + (size_t)NR * HD;
    f16* Vh = Kh + (size_t)NR * HD;
    f16* Chx = Vh + (size_t)NR * HD;

    qkv_kernel<<<dim3(128, 3), 256, 0, stream>>>(x, Wq, bq, Wk, bk, Wv, bv,
                                                 Qh, Kh, Vh);
    attn_kernel<<<dim3(16, 64), 256, 0, stream>>>(Qh, Kh, Vh, Chx);
    oln_kernel<<<128, 256, 0, stream>>>(Chx, Wo, bo, x, gamma, beta, out);
}
